// EncoderLayer_60971355734800
// MI455X (gfx1250) — matrix operand order repaired and checked by reading, not compiled, not run on hardware
//
#include <hip/hip_runtime.h>
#include <hip/hip_bf16.h>

// ---------------------------------------------------------------------------
// CDNA5 (gfx1250) sparse 3D encoder/decoder.
// Convs = implicit GEMM on v_wmma_f32_16x16x32_f16 (fp16 in, fp32 acc):
//   wave tile = 16 voxels (M) x 16 out-channels (N), K = taps*Cin in 64-wide
//   chunks staged through per-wave LDS tiles (ds_store_b16 / ds_load_b128
//   fragment reads matching the ISA 16-bit A/B layouts).
// Weights are DMA'd once per block into LDS via the Tensor Data Mover
// (tensor_load_to_lds + s_wait_tensorcnt), removing per-chunk global re-reads.
// Resident activations/masks/BN stats stay fp32 (HBM-bound at 23.3 TB/s).
// ---------------------------------------------------------------------------

typedef __attribute__((ext_vector_type(16))) _Float16 v16h;
typedef __attribute__((ext_vector_type(8)))  float    v8f;
typedef __attribute__((ext_vector_type(4)))  float    f32x4;

union HV { f32x4 q[2]; v16h h; };

// ---------------------------------------------------------------------------
// Tensor Data Mover: 1-row 2D tile load of `nelem` fp32 words global -> LDS.
// D# per CDNA5 ISA: group0 = {count=1, lds_addr, global_addr,
// type=2}; group1 = {data_size=4B, tensor_dim0=nelem, tensor_dim1=1,
// tile_dim0=nelem, tile_dim1=1, dim0_stride=nelem}; groups 2/3 zero (<=2D).
// ---------------------------------------------------------------------------
#if __has_builtin(__builtin_amdgcn_tensor_load_to_lds)
#define HAVE_TDM 1
typedef __attribute__((ext_vector_type(4))) unsigned u32x4;
typedef __attribute__((ext_vector_type(8))) int      i32x8;
typedef __attribute__((ext_vector_type(4))) int      i32x4;

__device__ __forceinline__ void tdm_load_f32_1d(float* lds_dst,
                                                const float* gsrc,
                                                unsigned nelem)
{
  const unsigned lds_off = (unsigned)(uintptr_t)lds_dst;   // LDS byte offset
  const unsigned long long ga = (unsigned long long)(uintptr_t)gsrc;
  u32x4 g0;
  g0.x = 1u;                                               // count=1, user D#
  g0.y = lds_off;                                          // lds_addr
  g0.z = (unsigned)ga;                                     // global_addr[31:0]
  g0.w = (unsigned)((ga >> 32) & 0x1FFFFFFu) | (2u << 30); // [56:32] | type=2
  i32x8 g1;
  g1[0] = 0x20000;                                         // data_size=2 (4B)
  g1[1] = (int)((nelem & 0xFFFFu) << 16);                  // tensor_dim0[15:0]
  g1[2] = (int)((nelem >> 16) | (1u << 16));               // td0[31:16]|td1=1
  g1[3] = (int)(nelem << 16);                              // tile_dim0
  g1[4] = 1;                                               // tile_dim1=1
  g1[5] = (int)nelem;                                      // dim0_stride
  g1[6] = 0; g1[7] = 0;
  i32x4 gz = {0, 0, 0, 0};
#if __clang_major__ >= 23
  i32x8 g4 = {0, 0, 0, 0, 0, 0, 0, 0};
  __builtin_amdgcn_tensor_load_to_lds(g0, g1, gz, gz, g4, 0);
#else
  __builtin_amdgcn_tensor_load_to_lds(g0, g1, gz, gz, 0);
#endif
#if __has_builtin(__builtin_amdgcn_s_wait_tensorcnt)
  __builtin_amdgcn_s_wait_tensorcnt(0);
#endif
}
#endif  // HAVE_TDM

// Stage weights into LDS: wave 0 issues the TDM DMA and drains TENSORcnt,
// then a workgroup barrier publishes the tile to all 4 waves.
__device__ __forceinline__ void stage_weights(float* Wlds, const float* w,
                                              int nelem)
{
#ifdef HAVE_TDM
  if ((threadIdx.x >> 5) == 0) tdm_load_f32_1d(Wlds, w, (unsigned)nelem);
#else
  for (int i = threadIdx.x; i < nelem; i += 128) Wlds[i] = w[i];
#endif
  __syncthreads();
}

// A-matrix fragment (16x32 fp16): lanes 0-15 hold K 0-7 & 16-23 of row M=lane,
// lanes 16-31 hold K 8-15 & 24-31 of row M=lane-16.
__device__ __forceinline__ v16h frag_a(const _Float16* row, int kkb, int lane) {
  const int ko = (lane < 16) ? 0 : 8;
  HV u;
  u.q[0] = *(const f32x4*)(row + kkb + ko);
  u.q[1] = *(const f32x4*)(row + kkb + 16 + ko);
  return u.h;
}

// B-matrix fragment (32x16 fp16): lanes 0-15 hold K 0-7 & 16-23 of col N=lane,
// lanes 16-31 hold K 8-15 & 24-31 of col N=lane-16.  LDS layout is [N][Krow].
__device__ __forceinline__ v16h frag_b(const _Float16* row, int kkb, int lane) {
  const int ko = (lane < 16) ? 0 : 8;
  const _Float16* p = row + kkb + ko;
  HV u;
  u.q[0] = *(const f32x4*)p;
  u.q[1] = *(const f32x4*)(p + 16);
  return u.h;
}

// ---------------------------------------------------------------------------
// Generic implicit-GEMM convolution.
//   out[v, n] = sum_{tap, c} in[neigh(v,tap), c] * w[tap, c, n]      (masked)
// Block = 128 thr = 4 waves, 64 out-voxels.  Weights TDM-staged in LDS when
// they fit in 32KB (every layer except the k=4 16->16 conv).
// ---------------------------------------------------------------------------
template<int KS, int STRIDE, int PADLO, int SIN, int SOUT,
         int CI, int CO, int CIP, int COP>
__global__ void __launch_bounds__(128) conv_wmma(
    const float* __restrict__ xin, const float* __restrict__ w,
    const float* __restrict__ maskOut, float* __restrict__ xout)
{
  constexpr int KS3  = KS * KS * KS;
  constexpr int KTOT = KS3 * CI;
  constexpr int KCH  = 64;
  constexpr int NKC  = (KTOT + KCH - 1) / KCH;
  constexpr int NT   = (COP + 15) / 16;
  constexpr int SO3  = SOUT * SOUT * SOUT;
  constexpr bool USE_WLDS = ((size_t)KTOT * CO <= 8192);   // <=32KB fp32

  __shared__ __align__(16) _Float16 As[4][16][KCH];
  __shared__ __align__(16) _Float16 Bs[4][16][KCH];
  __shared__ __align__(16) float Wlds[USE_WLDS ? (size_t)KTOT * CO : 4];

  const int wid   = threadIdx.x >> 5;
  const int lane  = threadIdx.x & 31;
  const int vbase = blockIdx.x * 64 + wid * 16;   // global out-voxel index
  const int b     = vbase / SO3;
  const int v0    = vbase % SO3;

  if (threadIdx.x == 0) __builtin_prefetch(xin, 0, 1);  // global_prefetch_b8
  if constexpr (USE_WLDS) stage_weights(Wlds, w, KTOT * CO);

  v8f acc[NT];
#pragma unroll
  for (int t = 0; t < NT; ++t)
#pragma unroll
    for (int r = 0; r < 8; ++r) acc[t][r] = 0.f;

  // per-lane im2col staging coordinates (lane fills 32 K-slots of one M row)
  const int am  = lane >> 1;
  const int ak0 = (lane & 1) * 32;
  const int avm = v0 + am;
  const int aoz = avm / (SOUT * SOUT);
  const int aoy = (avm / SOUT) % SOUT;
  const int aox = avm % SOUT;

  for (int kc = 0; kc < NKC; ++kc) {
    // ---- stage A chunk: As[m][kk] = im2col(voxel m, K = kc*64+kk) ----
#pragma unroll
    for (int e = 0; e < 32; ++e) {
      const int kk = ak0 + e;
      const int k  = kc * KCH + kk;
      float val = 0.f;
      if (k < KTOT) {
        const int tap = k / CI, c = k % CI;
        const int dz = tap / (KS * KS), dy = (tap / KS) % KS, dx = tap % KS;
        const int iz = aoz * STRIDE + dz - PADLO;
        const int iy = aoy * STRIDE + dy - PADLO;
        const int ix = aox * STRIDE + dx - PADLO;
        if ((unsigned)iz < (unsigned)SIN && (unsigned)iy < (unsigned)SIN &&
            (unsigned)ix < (unsigned)SIN)
          val = xin[((size_t)((b * SIN + iz) * SIN + iy) * SIN + ix) * CIP + c];
      }
      As[wid][am][kk] = (_Float16)val;
    }

    for (int nt = 0; nt < NT; ++nt) {
      // ---- stage B chunk: Bs[n][kk] = w[k, n_global] (k = tap*CI + c) ----
      const int bn  = lane >> 1;
      const int bng = nt * 16 + bn;
      const int bk0 = (lane & 1) * 32;
#pragma unroll
      for (int e = 0; e < 32; ++e) {
        const int kk = bk0 + e;
        const int k  = kc * KCH + kk;
        float val = 0.f;
        if (k < KTOT && bng < CO)
          val = USE_WLDS ? Wlds[k * CO + bng] : w[(size_t)k * CO + bng];
        Bs[wid][bn][kk] = (_Float16)val;
      }
      // ---- two K=32 WMMAs per 64-chunk (skip uniformly past K padding) ----
#pragma unroll
      for (int h = 0; h < 2; ++h) {
        if (kc * KCH + h * 32 < KTOT) {
          v16h a  = frag_a(&As[wid][lane & 15][0], h * 32, lane);
          v16h bf = frag_b(&Bs[wid][0][0] + (size_t)(lane & 15) * KCH, h * 32, lane);
          acc[nt] = __builtin_amdgcn_wmma_f32_16x16x32_f16(
              false, a, false, bf, (short)0, acc[nt], false, false);
        }
      }
    }
  }

  // ---- store D tile: lane holds N = lane&15, rows M = (lane<16?0:8)+r ----
  const int n0   = lane & 15;
  const int mrow = (lane < 16) ? 0 : 8;
#pragma unroll
  for (int r = 0; r < 8; ++r) {
    const int M = mrow + r;
    const float mval = maskOut[(size_t)b * SO3 + v0 + M];
#pragma unroll
    for (int nt = 0; nt < NT; ++nt) {
      const int n = nt * 16 + n0;
      if (n < COP) {
        const float val = (n < CO) ? acc[nt][r] * mval : 0.f;  // zero pad-chs
        xout[(size_t)(vbase + M) * COP + n] = val;
      }
    }
  }
}

// ---------------------------------------------------------------------------
// Transposed conv k=2 s=2 (FullConvolution): each fine voxel has exactly one
// parent, tap = parity.  Wave tile: 16 coarse voxels; A staged once, 8 parity
// GEMMs (K = Cin padded to 32) scatter to the 8 fine-voxel offsets, masked by
// the coarse mask (== up-sampled fine mask).  wu (<=16KB) is TDM-staged and
// reused by all 8 parity GEMMs.
// ---------------------------------------------------------------------------
template<int SC, int CI, int CO, int CIP, int COP>
__global__ void __launch_bounds__(128) convt_wmma(
    const float* __restrict__ xin, const float* __restrict__ wu,
    const float* __restrict__ maskC, float* __restrict__ xout)
{
  constexpr int SF  = SC * 2;
  constexpr int SC3 = SC * SC * SC;

  __shared__ __align__(16) _Float16 As[4][16][32];
  __shared__ __align__(16) _Float16 Bs[4][16][32];
  __shared__ __align__(16) float Wlds[8 * CI * CO];

  const int wid   = threadIdx.x >> 5;
  const int lane  = threadIdx.x & 31;
  const int vbase = blockIdx.x * 64 + wid * 16;
  const int b     = vbase / SC3;
  const int v0    = vbase % SC3;

  stage_weights(Wlds, wu, 8 * CI * CO);

  {
    const int m = lane >> 1, k0 = (lane & 1) * 16;
#pragma unroll
    for (int e = 0; e < 16; ++e) {
      const int kk = k0 + e;
      float val = (kk < CI) ? xin[(size_t)(vbase + m) * CIP + kk] : 0.f;
      As[wid][m][kk] = (_Float16)val;
    }
  }

  const int n0 = lane & 15;
  const int mrow = (lane < 16) ? 0 : 8;

  for (int p = 0; p < 8; ++p) {
    {
      const int n = lane >> 1, k0 = (lane & 1) * 16;
#pragma unroll
      for (int e = 0; e < 16; ++e) {
        const int kk = k0 + e;
        float val = (kk < CI && n < CO) ? Wlds[(p * CI + kk) * CO + n] : 0.f;
        Bs[wid][n][kk] = (_Float16)val;
      }
    }
    v8f acc;
#pragma unroll
    for (int r = 0; r < 8; ++r) acc[r] = 0.f;
    v16h a  = frag_a(&As[wid][lane & 15][0], 0, lane);
    v16h bf = frag_b(&Bs[wid][0][0] + (size_t)(lane & 15) * 32, 0, lane);
    acc = __builtin_amdgcn_wmma_f32_16x16x32_f16(
        false, a, false, bf, (short)0, acc, false, false);

    const int dz = (p >> 2) & 1, dy = (p >> 1) & 1, dx = p & 1;
#pragma unroll
    for (int r = 0; r < 8; ++r) {
      const int M = mrow + r;
      const int vm = v0 + M;
      const int cz = vm / (SC * SC), cy = (vm / SC) % SC, cx = vm % SC;
      const int fz = 2 * cz + dz, fy = 2 * cy + dy, fx = 2 * cx + dx;
      const float mval = maskC[(size_t)b * SC3 + vm];
      const size_t o = ((size_t)((b * SF + fz) * SF + fy) * SF + fx) * COP + n0;
      if (n0 < COP) xout[o] = (n0 < CO) ? acc[r] * mval : 0.f;
    }
  }
}

// ---------------------------------------------------------------------------
// Masked BatchNorm(+ReLU): grid-stride partial sums -> global f32 atomics,
// then elementwise apply.  stats[0..31]=sum, [32..63]=sumsq, [64]=mask count.
// (x is already masked everywhere, so plain sums equal masked sums.)
// ---------------------------------------------------------------------------
template<int CI, int CP>
__global__ void bn_reduce(const float* __restrict__ x,
                          const float* __restrict__ mask, int nvox,
                          float* __restrict__ stats)
{
  float s[CI], q[CI], cnt = 0.f;
#pragma unroll
  for (int c = 0; c < CI; ++c) { s[c] = 0.f; q[c] = 0.f; }
  for (int v = blockIdx.x * blockDim.x + threadIdx.x; v < nvox;
       v += gridDim.x * blockDim.x) {
    cnt += mask[v];
#pragma unroll
    for (int c = 0; c < CI; ++c) {
      float t = x[(size_t)v * CP + c];
      s[c] += t; q[c] += t * t;
    }
  }
#pragma unroll
  for (int c = 0; c < CI; ++c) {
    atomicAdd(&stats[c], s[c]);
    atomicAdd(&stats[32 + c], q[c]);
  }
  atomicAdd(&stats[64], cnt);
}

template<int CI, int CP>
__global__ void bn_apply(const float* __restrict__ x,
                         const float* __restrict__ mask,
                         const float* __restrict__ stats,
                         const float* __restrict__ g,
                         const float* __restrict__ bb,
                         float* __restrict__ y, int nvox)
{
  const int idx = blockIdx.x * blockDim.x + threadIdx.x;
  if (idx >= nvox * CP) return;
  const int v = idx / CP, c = idx % CP;
  float out = 0.f;
  if (c < CI) {
    const float cnt  = fmaxf(stats[64], 1.f);
    const float mean = stats[c] / cnt;
    const float var  = stats[32 + c] / cnt - mean * mean;
    const float t = (x[idx] - mean) * rsqrtf(var + 1e-4f) * g[c] + bb[c];
    out = fmaxf(t, 0.f) * mask[v];
  }
  y[idx] = out;
}

// ---------------------------------------------------------------------------
// Mask ops / elementwise glue
// ---------------------------------------------------------------------------
template<int SC>   // coarse size; reads fine 2*SC
__global__ void pool_mask(const float* __restrict__ mf, float* __restrict__ mc)
{
  constexpr int SF = 2 * SC, SC3 = SC * SC * SC;
  const int idx = blockIdx.x * blockDim.x + threadIdx.x;
  if (idx >= 128 * SC3) return;
  const int b = idx / SC3, v = idx % SC3;
  const int z = v / (SC * SC), y = (v / SC) % SC, x = v % SC;
  float m = 0.f;
#pragma unroll
  for (int dz = 0; dz < 2; ++dz)
#pragma unroll
    for (int dy = 0; dy < 2; ++dy)
#pragma unroll
      for (int dx = 0; dx < 2; ++dx)
        m = fmaxf(m, mf[((size_t)(b * SF + 2 * z + dz) * SF + (2 * y + dy)) * SF
                        + (2 * x + dx)]);
  mc[idx] = m;
}

template<int SF>   // fine size; reads coarse SF/2
__global__ void up_mask(const float* __restrict__ mc, float* __restrict__ mf)
{
  constexpr int SC = SF / 2, SF3 = SF * SF * SF;
  const int idx = blockIdx.x * blockDim.x + threadIdx.x;
  if (idx >= 128 * SF3) return;
  const int b = idx / SF3, v = idx % SF3;
  const int z = v / (SF * SF), y = (v / SF) % SF, x = v % SF;
  mf[idx] = mc[((size_t)(b * SC + z / 2) * SC + y / 2) * SC + x / 2];
}

template<int CP>
__global__ void sparsify(float* __restrict__ x, float* __restrict__ mask,
                         int nvox)
{
  const int v = blockIdx.x * blockDim.x + threadIdx.x;
  if (v >= nvox) return;
  const float m = mask[v] * ((x[(size_t)v * CP] > 0.f) ? 1.f : 0.f);
  mask[v] = m;
#pragma unroll
  for (int c = 0; c < CP; ++c) x[(size_t)v * CP + c] *= m;
}

__global__ void init0(const float* __restrict__ vox, const int* __restrict__ mraw,
                      float* __restrict__ mask, float* __restrict__ x, int nvox)
{
  const int v = blockIdx.x * blockDim.x + threadIdx.x;
  if (v >= nvox) return;
  const float m = (mraw[v] == 0) ? 1.f : 0.f;
  mask[v] = m;
  x[(size_t)v * 16] = vox[v] * m;
#pragma unroll
  for (int c = 1; c < 16; ++c) x[(size_t)v * 16 + c] = 0.f;
}

__global__ void hidden_extract(const float* __restrict__ x3,
                               float* __restrict__ out)
{
  const int idx = blockIdx.x * blockDim.x + threadIdx.x;
  if (idx >= 128 * 2048) return;
  const int b = idx >> 11, r = idx & 2047, c = r >> 6, v = r & 63;
  out[idx] = x3[((size_t)b * 64 + v) * 32 + c];   // NCDHW flatten
}

__global__ void zero_k(float* __restrict__ p, int n)
{
  const int i = blockIdx.x * blockDim.x + threadIdx.x;
  if (i < n) p[i] = 0.f;
}

// ---------------------------------------------------------------------------
extern "C" void kernel_launch(void* const* d_in, const int* in_sizes, int n_in,
                              void* d_out, int out_size, void* d_ws, size_t ws_size,
                              hipStream_t stream)
{
  (void)in_sizes; (void)n_in; (void)out_size; (void)ws_size;

  const float* vox  = (const float*)d_in[0];
  const int*   mraw = (const int*)d_in[1];
  const float* wp   = (const float*)d_in[2];
  const float *ge[3], *be[3], *wse[3], *wde[3];
  for (int i = 0; i < 3; ++i) {
    ge[i]  = (const float*)d_in[3 + 4 * i + 0];
    be[i]  = (const float*)d_in[3 + 4 * i + 1];
    wse[i] = (const float*)d_in[3 + 4 * i + 2];
    wde[i] = (const float*)d_in[3 + 4 * i + 3];
  }
  const float *gd[3], *bd[3], *wud[3], *ws3d[3], *ws4d[3];
  for (int j = 0; j < 3; ++j) {
    gd[j]   = (const float*)d_in[15 + 5 * j + 0];
    bd[j]   = (const float*)d_in[15 + 5 * j + 1];
    wud[j]  = (const float*)d_in[15 + 5 * j + 2];
    ws3d[j] = (const float*)d_in[15 + 5 * j + 3];
    ws4d[j] = (const float*)d_in[15 + 5 * j + 4];
  }
  const float* wo = (const float*)d_in[30];

  float* hidden = (float*)d_out;            // 128*2048
  float* outv   = (float*)d_out + 262144;   // 128*32^3

  // workspace carve (floats)
  float* W = (float*)d_ws;
  size_t o = 0;
  auto carve = [&](size_t n) { float* p = W + o; o += n; return p; };
  const int n0 = 128 * 32768, n1 = 128 * 4096, n2 = 128 * 512, n3 = 128 * 64;
  float* b0a = carve((size_t)n0 * 16);
  float* b0b = carve((size_t)n0 * 16);
  float* b1a = carve((size_t)n1 * 16);
  float* b1b = carve((size_t)n1 * 16);
  float* b2a = carve((size_t)n2 * 16);
  float* b2b = carve((size_t)n2 * 16);
  float* b3a = carve((size_t)n3 * 32);
  float* b3b = carve((size_t)n3 * 32);
  float* m0 = carve(n0);
  float* m1 = carve(n1);
  float* m2 = carve(n2);
  float* m3 = carve(n3);
  float* st = carve(80);

  auto g1 = [](long n, int t) { return (unsigned)((n + t - 1) / t); };

  // ---- prepare ----
  init0<<<g1(n0, 256), 256, 0, stream>>>(vox, mraw, m0, b0a, n0);
  conv_wmma<3,1,1, 32,32, 1,4, 16,16><<<n0/64, 128, 0, stream>>>(b0a, wp, m0, b0b);

  // ---- encoder stage 0 (4ch @32 -> 8ch @16) ----
  zero_k<<<1, 128, 0, stream>>>(st, 80);
  bn_reduce<4,16><<<256, 256, 0, stream>>>(b0b, m0, n0, st);
  bn_apply<4,16><<<g1((long)n0*16, 256), 256, 0, stream>>>(b0b, m0, st, ge[0], be[0], b0a, n0);
  conv_wmma<3,1,1, 32,32, 4,4, 16,16><<<n0/64, 128, 0, stream>>>(b0a, wse[0], m0, b0b);
  pool_mask<16><<<g1(n1, 256), 256, 0, stream>>>(m0, m1);
  conv_wmma<2,2,0, 32,16, 4,8, 16,16><<<n1/64, 128, 0, stream>>>(b0b, wde[0], m1, b1a);

  // ---- encoder stage 1 (8ch @16 -> 16ch @8) ----
  zero_k<<<1, 128, 0, stream>>>(st, 80);
  bn_reduce<8,16><<<256, 256, 0, stream>>>(b1a, m1, n1, st);
  bn_apply<8,16><<<g1((long)n1*16, 256), 256, 0, stream>>>(b1a, m1, st, ge[1], be[1], b1b, n1);
  conv_wmma<3,1,1, 16,16, 8,8, 16,16><<<n1/64, 128, 0, stream>>>(b1b, wse[1], m1, b1a);
  pool_mask<8><<<g1(n2, 256), 256, 0, stream>>>(m1, m2);
  conv_wmma<2,2,0, 16,8, 8,16, 16,16><<<n2/64, 128, 0, stream>>>(b1a, wde[1], m2, b2a);

  // ---- encoder stage 2 (16ch @8 -> 32ch @4) ----
  zero_k<<<1, 128, 0, stream>>>(st, 80);
  bn_reduce<16,16><<<256, 256, 0, stream>>>(b2a, m2, n2, st);
  bn_apply<16,16><<<g1((long)n2*16, 256), 256, 0, stream>>>(b2a, m2, st, ge[2], be[2], b2b, n2);
  conv_wmma<3,1,1, 8,8, 16,16, 16,16><<<n2/64, 128, 0, stream>>>(b2b, wse[2], m2, b2a);
  pool_mask<4><<<g1(n3, 256), 256, 0, stream>>>(m2, m3);
  conv_wmma<2,2,0, 8,4, 16,32, 16,32><<<n3/64, 128, 0, stream>>>(b2a, wde[2], m3, b3a);

  // ---- hidden ----
  hidden_extract<<<g1(128 * 2048, 256), 256, 0, stream>>>(b3a, hidden);

  // ---- decoder stage 0 (32ch @4 -> 16ch @8) ----
  zero_k<<<1, 128, 0, stream>>>(st, 80);
  bn_reduce<32,32><<<256, 256, 0, stream>>>(b3a, m3, n3, st);
  bn_apply<32,32><<<g1((long)n3*32, 256), 256, 0, stream>>>(b3a, m3, st, gd[0], bd[0], b3b, n3);
  up_mask<8><<<g1(n2, 256), 256, 0, stream>>>(m3, m2);
  convt_wmma<4, 32,16, 32,16><<<n3/64, 128, 0, stream>>>(b3b, wud[0], m3, b2a);
  conv_wmma<3,1,1, 8,8, 16,16, 16,16><<<n2/64, 128, 0, stream>>>(b2a, ws3d[0], m2, b2b);
  sparsify<16><<<g1(n2, 256), 256, 0, stream>>>(b2b, m2, n2);
  conv_wmma<4,1,1, 8,8, 16,16, 16,16><<<n2/64, 128, 0, stream>>>(b2b, ws4d[0], m2, b2a);

  // ---- decoder stage 1 (16ch @8 -> 8ch @16) ----
  zero_k<<<1, 128, 0, stream>>>(st, 80);
  bn_reduce<16,16><<<256, 256, 0, stream>>>(b2a, m2, n2, st);
  bn_apply<16,16><<<g1((long)n2*16, 256), 256, 0, stream>>>(b2a, m2, st, gd[1], bd[1], b2b, n2);
  up_mask<16><<<g1(n1, 256), 256, 0, stream>>>(m2, m1);
  convt_wmma<8, 16,8, 16,16><<<n2/64, 128, 0, stream>>>(b2b, wud[1], m2, b1a);
  conv_wmma<3,1,1, 16,16, 8,8, 16,16><<<n1/64, 128, 0, stream>>>(b1a, ws3d[1], m1, b1b);
  sparsify<16><<<g1(n1, 256), 256, 0, stream>>>(b1b, m1, n1);
  conv_wmma<4,1,1, 16,16, 8,8, 16,16><<<n1/64, 128, 0, stream>>>(b1b, ws4d[1], m1, b1a);

  // ---- decoder stage 2 (8ch @16 -> 4ch @32) ----
  zero_k<<<1, 128, 0, stream>>>(st, 80);
  bn_reduce<8,16><<<256, 256, 0, stream>>>(b1a, m1, n1, st);
  bn_apply<8,16><<<g1((long)n1*16, 256), 256, 0, stream>>>(b1a, m1, st, gd[2], bd[2], b1b, n1);
  up_mask<32><<<g1(n0, 256), 256, 0, stream>>>(m1, m0);
  convt_wmma<16, 8,4, 16,16><<<n1/64, 128, 0, stream>>>(b1b, wud[2], m1, b0a);
  conv_wmma<3,1,1, 32,32, 4,4, 16,16><<<n0/64, 128, 0, stream>>>(b0a, ws3d[2], m0, b0b);
  sparsify<16><<<g1(n0, 256), 256, 0, stream>>>(b0b, m0, n0);
  conv_wmma<4,1,1, 32,32, 4,4, 16,16><<<n0/64, 128, 0, stream>>>(b0b, ws4d[2], m0, b0a);

  // ---- output conv (4ch -> 1ch @32) ----
  conv_wmma<3,1,1, 32,32, 4,1, 16,1><<<n0/64, 128, 0, stream>>>(b0a, wo, m0, outv);
}
